// SemiCompressedAttention_84310208021037
// MI455X (gfx1250) — matrix operand order repaired and checked by reading, not compiled, not run on hardware
//
#include <hip/hip_runtime.h>

typedef __attribute__((ext_vector_type(16))) _Float16 v16h;
typedef __attribute__((ext_vector_type(8)))  float    v8f;

// ---------------- problem constants ----------------
constexpr int Bc  = 2;
constexpr int Tc  = 2048;
constexpr int Dc  = 1024;
constexpr int Hc  = 4;
constexpr int Wc  = 512;
constexpr int Sc  = 64;
constexpr int CKc = Dc / Hc;   // 256
constexpr int CVc = Dc / Hc;   // 256
constexpr int HSc = Hc * Sc;   // 256
constexpr float EPSc = 1e-5f;

// ---------------- WMMA helpers ----------------
__device__ inline v8f wmma16(v16h a, v16h b, v8f c) {
  return __builtin_amdgcn_wmma_f32_16x16x32_f16(false, a, false, b, (short)0, c,
                                                false, false);
}

// A-fragment (16x32, f16 row-major, ld in halves).
// Lane l: m = l&15; lanes 0-15 hold K 0-15, lanes 16-31 hold K 16-31 (contiguous).
__device__ inline v16h frag_a_h(const _Float16* __restrict__ base, int ld) {
  const int lane = threadIdx.x & 31;
  const _Float16* row = base + (size_t)(lane & 15) * ld + ((lane >> 4) << 4);
  v16h f;
#pragma unroll
  for (int i = 0; i < 8; ++i) f[i] = row[i];
#pragma unroll
  for (int i = 0; i < 8; ++i) f[8 + i] = row[8 + i];
  return f;
}

// Same A-fragment but sourced from f32 (LDS score panel), converting to f16.
__device__ inline v16h frag_a_f(const float* __restrict__ base, int ld) {
  const int lane = threadIdx.x & 31;
  const float* row = base + (size_t)(lane & 15) * ld + ((lane >> 4) << 4);
  v16h f;
#pragma unroll
  for (int i = 0; i < 8; ++i) f[i] = (_Float16)row[i];
#pragma unroll
  for (int i = 0; i < 8; ++i) f[8 + i] = (_Float16)row[8 + i];
  return f;
}

// B-fragment from an N x K row-major array (operand is mathematically K x N = src^T).
// Lane l: n = l&15; lanes 0-15 hold K 0-15, lanes 16-31 hold K 16-31 (contiguous).
__device__ inline v16h frag_b_nk(const _Float16* __restrict__ base, int ld) {
  const int lane = threadIdx.x & 31;
  const _Float16* row = base + (size_t)(lane & 15) * ld + ((lane >> 4) << 4);
  v16h f;
#pragma unroll
  for (int i = 0; i < 16; ++i) f[i] = row[i];
  return f;
}

// N x K B-fragment with row clamping (rows may be out of [0,T)); invalid rows are
// masked to zero probability upstream so clamped garbage contributes nothing.
__device__ inline v16h frag_b_nk_clamp(const _Float16* __restrict__ origin, int ld,
                                       int rowbase, int rowmax) {
  const int lane = threadIdx.x & 31;
  int r = rowbase + (lane & 15);
  r = r < 0 ? 0 : (r >= rowmax ? rowmax - 1 : r);
  const _Float16* row = origin + (size_t)r * ld + ((lane >> 4) << 4);
  v16h f;
#pragma unroll
  for (int i = 0; i < 16; ++i) f[i] = row[i];
  return f;
}

// B-fragment from a K x N row-major array (operand used directly, K = rows).
__device__ inline v16h frag_b_kn(const _Float16* __restrict__ base, int ld) {
  const int lane = threadIdx.x & 31;
  const _Float16* col = base + (lane & 15) + (size_t)((lane >> 4) << 4) * ld;
  v16h f;
#pragma unroll
  for (int i = 0; i < 16; ++i) f[i] = col[(size_t)i * ld];
  return f;
}

// K x N B-fragment with row clamping.
__device__ inline v16h frag_b_kn_clamp(const _Float16* __restrict__ origin, int ld,
                                       int rowbase, int rowmax) {
  const int lane = threadIdx.x & 31;
  const int n = lane & 15, klo = (lane >> 4) << 4;
  v16h f;
#pragma unroll
  for (int i = 0; i < 16; ++i) {
    int r = rowbase + klo + i;
    r = r < 0 ? 0 : (r >= rowmax ? rowmax - 1 : r);
    f[i] = origin[(size_t)r * ld + n];
  }
  return f;
}

// CDNA5 async copy: 16 bytes per lane, global -> LDS, tracked by ASYNCcnt.
// IOFFSET is applied to BOTH the LDS and the global address (ISA 15.18.3 op 98).
__device__ inline void async_b128(uint32_t lds_off, const void* gaddr) {
  asm volatile("global_load_async_to_lds_b128 %0, %1, off"
               :: "v"(lds_off), "v"((uint64_t)gaddr) : "memory");
}
__device__ inline void async_b128_o16(uint32_t lds_off, const void* gaddr) {
  asm volatile("global_load_async_to_lds_b128 %0, %1, off offset:16"
               :: "v"(lds_off), "v"((uint64_t)gaddr) : "memory");
}
__device__ inline void wait_async0() {
  asm volatile("s_wait_asynccnt 0x0" ::: "memory");
}

// ---------------- elementwise kernels ----------------
__global__ __launch_bounds__(256) void cast_f32_to_f16(
    const float* __restrict__ in, _Float16* __restrict__ out, int n) {
  int i = blockIdx.x * 256 + threadIdx.x;
  if (i < n) out[i] = (_Float16)in[i];
}

// RMS-norm one D=1024 row, cast to f16. grid = B*T, block = 256.
__global__ __launch_bounds__(256) void rms_cast(
    const float* __restrict__ x, const float* __restrict__ wn,
    _Float16* __restrict__ out) {
  __shared__ float rbuf[256];
  const int row = blockIdx.x, tid = threadIdx.x;
  const float* xr = x + (size_t)row * Dc;
  float l[4], ss = 0.f;
#pragma unroll
  for (int i = 0; i < 4; ++i) { l[i] = xr[tid + 256 * i]; ss += l[i] * l[i]; }
  rbuf[tid] = ss;
  __syncthreads();
  for (int s = 128; s > 0; s >>= 1) {
    if (tid < s) rbuf[tid] += rbuf[tid + s];
    __syncthreads();
  }
  const float inv = rsqrtf(rbuf[0] / (float)Dc + EPSc);
#pragma unroll
  for (int i = 0; i < 4; ++i) {
    int c = tid + 256 * i;
    out[(size_t)row * Dc + c] = (_Float16)(l[i] * inv * wn[c]);
  }
}

// SiLU then RMS-norm one row, cast to f16. grid = B*T, block = 256.
__global__ __launch_bounds__(256) void silu_rms_cast(
    const float* __restrict__ x, const float* __restrict__ wn,
    _Float16* __restrict__ out) {
  __shared__ float rbuf[256];
  const int row = blockIdx.x, tid = threadIdx.x;
  const float* xr = x + (size_t)row * Dc;
  float l[4], ss = 0.f;
#pragma unroll
  for (int i = 0; i < 4; ++i) {
    float v = xr[tid + 256 * i];
    float sv = v / (1.f + __expf(-v));
    l[i] = sv;
    ss += sv * sv;
  }
  rbuf[tid] = ss;
  __syncthreads();
  for (int s = 128; s > 0; s >>= 1) {
    if (tid < s) rbuf[tid] += rbuf[tid + s];
    __syncthreads();
  }
  const float inv = rsqrtf(rbuf[0] / (float)Dc + EPSc);
#pragma unroll
  for (int i = 0; i < 4; ++i) {
    int c = tid + 256 * i;
    out[(size_t)row * Dc + c] = (_Float16)(l[i] * inv * wn[c]);
  }
}

// softmax over 256-wide rows of g. grid = B*T, block = 256.
__global__ __launch_bounds__(256) void softmax_g(
    const _Float16* __restrict__ gin, _Float16* __restrict__ gout) {
  __shared__ float rb[256];
  const int row = blockIdx.x, tid = threadIdx.x;
  const float x = (float)gin[(size_t)row * HSc + tid];
  rb[tid] = x;
  __syncthreads();
  for (int s = 128; s > 0; s >>= 1) {
    if (tid < s) rb[tid] = fmaxf(rb[tid], rb[tid + s]);
    __syncthreads();
  }
  const float mx = rb[0];
  __syncthreads();
  const float e = __expf(x - mx);
  rb[tid] = e;
  __syncthreads();
  for (int s = 128; s > 0; s >>= 1) {
    if (tid < s) rb[tid] += rb[tid + s];
    __syncthreads();
  }
  gout[(size_t)row * HSc + tid] = (_Float16)(e / rb[0]);
}

// ---------------- WMMA GEMM: C[M,N] = A[M,K] (f16) * W[N,K]^T (f16) ----------------
// block = 128 threads (4 waves), tile 64x64, K-step 32.
// Tiles staged with global_load_async_to_lds_b128 (ASYNCcnt) + global_prefetch_b8.
#define GLD 40  // LDS row stride in halves (32 + 8 pad); 80 B = 5*16 B aligned
__global__ __launch_bounds__(128) void gemm_f16(
    const _Float16* __restrict__ A, const _Float16* __restrict__ Wt,
    float* __restrict__ Cf, _Float16* __restrict__ Ch, int M, int N, int K) {
  __shared__ _Float16 As[64 * GLD];
  __shared__ _Float16 Bs[64 * GLD];
  const int tid = threadIdx.x, lane = tid & 31, wave = tid >> 5;
  const int wm = (wave >> 1) * 32, wn = (wave & 1) * 32;
  const size_t m0 = (size_t)blockIdx.x * 64, n0 = (size_t)blockIdx.y * 64;
  const int row = tid >> 1, seg = (tid & 1) * 16;  // each lane owns 32 B of a row

  const uint32_t ldsA = (uint32_t)(size_t)(As + row * GLD + seg);
  const uint32_t ldsB = (uint32_t)(size_t)(Bs + row * GLD + seg);
  const _Float16* gA = A + (m0 + row) * K + seg;
  const _Float16* gB = Wt + (n0 + row) * K + seg;

  v8f acc[2][2];
#pragma unroll
  for (int i = 0; i < 2; ++i)
#pragma unroll
    for (int j = 0; j < 2; ++j) acc[i][j] = {};

  for (int k0 = 0; k0 < K; k0 += 32) {
    __syncthreads();  // all waves done reading the previous tile
    async_b128(ldsA, gA + k0);
    async_b128_o16(ldsA, gA + k0);
    async_b128(ldsB, gB + k0);
    async_b128_o16(ldsB, gB + k0);
    if (k0 + 32 < K) {  // global_prefetch_b8 the next K tile
      __builtin_prefetch(gA + k0 + 32, 0, 1);
      __builtin_prefetch(gB + k0 + 32, 0, 1);
    }
    wait_async0();      // this wave's async copies landed in LDS
    __syncthreads();    // every wave's copies landed
    v16h a0 = frag_a_h(As + (wm + 0) * GLD, GLD);
    v16h a1 = frag_a_h(As + (wm + 16) * GLD, GLD);
    v16h b0 = frag_b_nk(Bs + (wn + 0) * GLD, GLD);
    v16h b1 = frag_b_nk(Bs + (wn + 16) * GLD, GLD);
    acc[0][0] = wmma16(a0, b0, acc[0][0]);
    acc[0][1] = wmma16(a0, b1, acc[0][1]);
    acc[1][0] = wmma16(a1, b0, acc[1][0]);
    acc[1][1] = wmma16(a1, b1, acc[1][1]);
  }

  const int n = lane & 15, hi = lane >> 4;
  if (Cf) {
#pragma unroll
    for (int i = 0; i < 2; ++i)
#pragma unroll
      for (int j = 0; j < 2; ++j)
#pragma unroll
        for (int r = 0; r < 8; ++r)
          Cf[(m0 + wm + 16 * i + r + 8 * hi) * N + (n0 + wn + 16 * j + n)] =
              acc[i][j][r];
  }
  if (Ch) {
#pragma unroll
    for (int i = 0; i < 2; ++i)
#pragma unroll
      for (int j = 0; j < 2; ++j)
#pragma unroll
        for (int r = 0; r < 8; ++r)
          Ch[(m0 + wm + 16 * i + r + 8 * hi) * N + (n0 + wn + 16 * j + n)] =
              (_Float16)acc[i][j][r];
  }
}

// ---------------- fused attention ----------------
// grid = (T/16, H, B); block = 128 (4 waves). One block owns a 16-row tile.
#define SCW 608   // 64 state cols + 544 window cols (528 used + pad)
__global__ __launch_bounds__(128) void attn_fused(
    const _Float16* __restrict__ q16, const _Float16* __restrict__ k16,
    const _Float16* __restrict__ v16, const _Float16* __restrict__ s16,
    const _Float16* __restrict__ g16, float* __restrict__ obuf) {
  __shared__ float sc[16][SCW];            // joint score / prob panel (f32)
  __shared__ _Float16 a16[4][16][40];      // wave-private f16 relay tiles
  __shared__ float red[16][8];             // softmax reduction scratch

  const int t0 = blockIdx.x * 16;
  const int h  = blockIdx.y;
  const int bb = blockIdx.z;
  const int tid = threadIdx.x, lane = tid & 31, w = tid >> 5;
  const int n = lane & 15, hi = lane >> 4;

  const size_t bofs = (size_t)bb * Tc * Dc;
  const _Float16* qh = q16 + bofs + (size_t)h * CKc;
  const _Float16* kh = k16 + bofs + (size_t)h * CKc;
  const _Float16* vh = v16 + bofs + (size_t)h * CVc;
  const _Float16* sh = s16 + bofs + (size_t)h * CKc;
  const _Float16* gh = g16 + (size_t)bb * Tc * HSc + (size_t)h * Sc;

  // zero the state-score columns (reduced into with ds_add_f32)
  for (int i = tid; i < 16 * Sc; i += 128) sc[i >> 6][i & 63] = 0.f;
  __syncthreads();

  // preload q row-tile fragments (16 x 256 -> 8 K-chunks)
  v16h qa[8];
#pragma unroll
  for (int kc = 0; kc < 8; ++kc)
    qa[kc] = frag_a_h(qh + (size_t)t0 * Dc + 32 * kc, Dc);

  const int T16 = t0 / 16 + 1;           // causal u-tiles
  const int npairs = (T16 + 1) >> 1;     // K=32 pairs of u-tiles

  // ---- pass 1a: sc_state += mask(q s^T) * g, u-tile pairs strided over waves ----
  v8f scacc[4];
#pragma unroll
  for (int i = 0; i < 4; ++i) scacc[i] = {};
  for (int p = w; p < npairs; p += 4) {
    const int ubase = 32 * p;
#pragma unroll
    for (int j = 0; j < 2; ++j) {
      const int u0 = ubase + 16 * j;
      if (u0 <= t0) {
        v8f aacc = {};
#pragma unroll
        for (int kc = 0; kc < 8; ++kc) {
          v16h bs_ = frag_b_nk(sh + (size_t)u0 * Dc + 32 * kc, Dc);  // s^T
          aacc = wmma16(qa[kc], bs_, aacc);
        }
#pragma unroll
        for (int r = 0; r < 8; ++r) {
          int tt = t0 + r + 8 * hi, uu = u0 + n;
          float vv = (uu <= tt) ? aacc[r] : 0.f;   // causal mask
          a16[w][r + 8 * hi][16 * j + n] = (_Float16)vv;
        }
      } else {
#pragma unroll
        for (int r = 0; r < 8; ++r) a16[w][r + 8 * hi][16 * j + n] = (_Float16)0.f;
      }
    }
    v16h fa = frag_a_h(&a16[w][0][0], 40);
#pragma unroll
    for (int nt = 0; nt < 4; ++nt) {
      v16h bg = frag_b_kn(gh + (size_t)ubase * HSc + 16 * nt, HSc);
      scacc[nt] = wmma16(fa, bg, scacc[nt]);
    }
  }
#pragma unroll
  for (int nt = 0; nt < 4; ++nt)
#pragma unroll
    for (int r = 0; r < 8; ++r)
      atomicAdd(&sc[r + 8 * hi][16 * nt + n], scacc[nt][r]);

  // ---- pass 1b: sliding-window scores q k^T + ALiBi, 33 u-tiles over waves ----
  const float slope = exp2f(-8.0f * (float)h / (float)Hc);
  for (int ti = w; ti < 33; ti += 4) {
    const int u0 = t0 - Wc + 16 * ti;
    if (u0 + 16 <= 0) {
#pragma unroll
      for (int r = 0; r < 8; ++r) sc[r + 8 * hi][64 + 16 * ti + n] = -1e30f;
      continue;
    }
    v8f aacc = {};
#pragma unroll
    for (int kc = 0; kc < 8; ++kc) {
      v16h bk = frag_b_nk_clamp(kh + 32 * kc, Dc, u0, Tc);  // k^T, clamped rows
      aacc = wmma16(qa[kc], bk, aacc);
    }
#pragma unroll
    for (int r = 0; r < 8; ++r) {
      int tt = t0 + r + 8 * hi, uu = u0 + n, d = tt - uu;
      float val = (uu >= 0 && d >= 0 && d < Wc) ? (aacc[r] - slope * (float)d)
                                                : -1e30f;
      sc[r + 8 * hi][64 + 16 * ti + n] = val;
    }
  }
  __syncthreads();

  // ---- joint softmax over 64 + 528 columns, in place ----
  {
    const int row = tid >> 3, sub = tid & 7;
    float m_ = -3.0e38f;
    for (int c = sub; c < 592; c += 8) m_ = fmaxf(m_, sc[row][c]);
    red[row][sub] = m_;
    __syncthreads();
    float rmax = red[row][0];
#pragma unroll
    for (int i = 1; i < 8; ++i) rmax = fmaxf(rmax, red[row][i]);
    __syncthreads();
    float s_ = 0.f;
    for (int c = sub; c < 592; c += 8) {
      float e = __expf(sc[row][c] - rmax);
      sc[row][c] = e;
      s_ += e;
    }
    red[row][sub] = s_;
    __syncthreads();
    float rsum = 0.f;
#pragma unroll
    for (int i = 0; i < 8; ++i) rsum += red[row][i];
    const float inv = 1.0f / rsum;
    for (int c = sub; c < 592; c += 8) sc[row][c] *= inv;
    for (int c = 592 + sub; c < SCW; c += 8) sc[row][c] = 0.f;  // pad -> p = 0
  }
  __syncthreads();

  // ---- pass 2: o = p_w v2 + mask(p_state g^T) v ; wave owns 64 output cols ----
  v16h pa0 = frag_a_f(&sc[0][0], SCW);
  v16h pa1 = frag_a_f(&sc[0][32], SCW);
  v8f oc[4];
#pragma unroll
  for (int i = 0; i < 4; ++i) oc[i] = {};

  // windowed part: 17 K=32 chunks across the 544-wide window panel
  for (int pr = 0; pr < 17; ++pr) {
    const int kb = 32 * pr;
    v16h fa = frag_a_f(&sc[0][64 + kb], SCW);
    const int rowbase = t0 - Wc + kb;
#pragma unroll
    for (int nt = 0; nt < 4; ++nt) {
      v16h bv = frag_b_kn_clamp(vh + 64 * w + 16 * nt, Dc, rowbase, Tc);
      oc[nt] = wmma16(fa, bv, oc[nt]);
    }
  }

  // state part: every wave walks all causal u-tile pairs (columns are split)
  for (int p = 0; p < npairs; ++p) {
    const int ubase = 32 * p;
#pragma unroll
    for (int j = 0; j < 2; ++j) {
      const int u0 = ubase + 16 * j;
      if (u0 <= t0) {
        v8f macc = {};
        v16h bg0 = frag_b_nk(gh + (size_t)u0 * HSc, HSc);        // g^T
        macc = wmma16(pa0, bg0, macc);
        v16h bg1 = frag_b_nk(gh + (size_t)u0 * HSc + 32, HSc);
        macc = wmma16(pa1, bg1, macc);
#pragma unroll
        for (int r = 0; r < 8; ++r) {
          int tt = t0 + r + 8 * hi, uu = u0 + n;
          float vv = (uu <= tt) ? macc[r] : 0.f;   // causal mask on M
          a16[w][r + 8 * hi][16 * j + n] = (_Float16)vv;
        }
      } else {
#pragma unroll
        for (int r = 0; r < 8; ++r) a16[w][r + 8 * hi][16 * j + n] = (_Float16)0.f;
      }
    }
    v16h fm = frag_a_h(&a16[w][0][0], 40);
#pragma unroll
    for (int nt = 0; nt < 4; ++nt) {
      v16h bv = frag_b_kn(vh + (size_t)ubase * Dc + 64 * w + 16 * nt, Dc);
      oc[nt] = wmma16(fm, bv, oc[nt]);
    }
  }

  // store o tile (f32)
#pragma unroll
  for (int nt = 0; nt < 4; ++nt)
#pragma unroll
    for (int r = 0; r < 8; ++r)
      obuf[(size_t)(bb * Tc + t0 + r + 8 * hi) * Dc + h * CVc + 64 * w + 16 * nt + n] =
          oc[nt][r];
}

// ---------------- launcher ----------------
extern "C" void kernel_launch(void* const* d_in, const int* in_sizes, int n_in,
                              void* d_out, int out_size, void* d_ws, size_t ws_size,
                              hipStream_t stream) {
  (void)in_sizes; (void)n_in; (void)out_size; (void)ws_size;
  const float* hidden = (const float*)d_in[0];
  const float* w_norm = (const float*)d_in[1];
  const float* Wq = (const float*)d_in[2];
  const float* Wk = (const float*)d_in[3];
  const float* Wv = (const float*)d_in[4];
  const float* Ws = (const float*)d_in[5];
  const float* Wg = (const float*)d_in[6];
  const float* Wo = (const float*)d_in[7];
  float* out = (float*)d_out;

  char* wsp = (char*)d_ws;
  auto carve = [&](size_t bytes) -> void* {
    void* r = (void*)wsp;
    wsp += (bytes + 255) & ~(size_t)255;
    return r;
  };
  const size_t DD   = (size_t)Dc * Dc;        // 1M
  const size_t BTD  = (size_t)Bc * Tc * Dc;   // 4M
  const size_t BTHS = (size_t)Bc * Tc * HSc;  // 1M

  _Float16* wq16 = (_Float16*)carve(DD * 2);
  _Float16* wk16 = (_Float16*)carve(DD * 2);
  _Float16* wv16 = (_Float16*)carve(DD * 2);
  _Float16* ws16 = (_Float16*)carve(DD * 2);
  _Float16* wo16 = (_Float16*)carve(DD * 2);
  _Float16* wg16 = (_Float16*)carve((size_t)HSc * Dc * 2);
  _Float16* h16  = (_Float16*)carve(BTD * 2);
  _Float16* q16  = (_Float16*)carve(BTD * 2);
  _Float16* k16  = (_Float16*)carve(BTD * 2);
  _Float16* v16b = (_Float16*)carve(BTD * 2);
  _Float16* s16  = (_Float16*)carve(BTD * 2);
  _Float16* graw = (_Float16*)carve(BTHS * 2);
  _Float16* g16  = (_Float16*)carve(BTHS * 2);
  float*    obuf = (float*)carve(BTD * 4);
  _Float16* o16  = (_Float16*)carve(BTD * 2);

  // weights -> f16
  cast_f32_to_f16<<<dim3((unsigned)((DD + 255) / 256)), 256, 0, stream>>>(Wq, wq16, (int)DD);
  cast_f32_to_f16<<<dim3((unsigned)((DD + 255) / 256)), 256, 0, stream>>>(Wk, wk16, (int)DD);
  cast_f32_to_f16<<<dim3((unsigned)((DD + 255) / 256)), 256, 0, stream>>>(Wv, wv16, (int)DD);
  cast_f32_to_f16<<<dim3((unsigned)((DD + 255) / 256)), 256, 0, stream>>>(Ws, ws16, (int)DD);
  cast_f32_to_f16<<<dim3((unsigned)((DD + 255) / 256)), 256, 0, stream>>>(Wo, wo16, (int)DD);
  cast_f32_to_f16<<<dim3((unsigned)((HSc * Dc + 255) / 256)), 256, 0, stream>>>(Wg, wg16, HSc * Dc);

  // RMS norm + cast
  rms_cast<<<Bc * Tc, 256, 0, stream>>>(hidden, w_norm, h16);

  // projections (WMMA GEMMs, async-LDS staged)
  const dim3 gproj(Bc * Tc / 64, Dc / 64);
  gemm_f16<<<gproj, 128, 0, stream>>>(h16, wq16, nullptr, q16, Bc * Tc, Dc, Dc);
  gemm_f16<<<gproj, 128, 0, stream>>>(h16, wk16, nullptr, k16, Bc * Tc, Dc, Dc);
  gemm_f16<<<gproj, 128, 0, stream>>>(h16, wv16, nullptr, v16b, Bc * Tc, Dc, Dc);
  gemm_f16<<<gproj, 128, 0, stream>>>(h16, ws16, nullptr, s16, Bc * Tc, Dc, Dc);
  gemm_f16<<<dim3(Bc * Tc / 64, HSc / 64), 128, 0, stream>>>(h16, wg16, nullptr, graw,
                                                             Bc * Tc, HSc, Dc);
  softmax_g<<<Bc * Tc, 256, 0, stream>>>(graw, g16);

  // fused attention
  attn_fused<<<dim3(Tc / 16, Hc, Bc), 128, 0, stream>>>(q16, k16, v16b, s16, g16, obuf);

  // SiLU + RMS + output projection
  silu_rms_cast<<<Bc * Tc, 256, 0, stream>>>(obuf, w_norm, o16);
  gemm_f16<<<dim3(Bc * Tc / 64, Dc / 64), 128, 0, stream>>>(o16, wo16, out, nullptr,
                                                            Bc * Tc, Dc, Dc);
}
